// AttentionModule_7146825580577
// MI455X (gfx1250) — matrix operand order repaired and checked by reading, not compiled, not run on hardware
//
#include <hip/hip_runtime.h>
#include <hip/hip_bf16.h>

// ---------------------------------------------------------------------------
// MI455X (gfx1250) implementation.
// All heavy math runs through v_wmma_f32_16x16x32_bf16 (wave32 WMMA, fp32
// accumulate). Weights are pre-transposed+cast to bf16 [N,K] so B-fragments
// are contiguous 32B loads; activations are cast to bf16 between stages.
// Each wave register-blocks a 32x32 output tile (2x2 WMMA tiles) so only
// 2 b128 loads are needed per WMMA issue.
// ---------------------------------------------------------------------------

typedef __bf16 bf16_t;
typedef __attribute__((ext_vector_type(16))) __bf16 v16bf;
typedef __attribute__((ext_vector_type(8)))  float          v8f;
typedef __attribute__((ext_vector_type(8)))  unsigned short u16x8;

// ---------------------------------------------------------------------------
// GEMM: C[M,N] = A[M,K](bf16) * Bt[N,K](bf16, pre-transposed) + bias
// Block: 256 threads = 8 waves arranged 4(M) x 2(N); each wave computes a
// 32x32 tile (2x2 WMMA tiles) -> block tile 128x64.
//
// Fragment layouts used here (A/B in contiguous K order, not the ISA table order):
//  A (16x32 bf16): lanes 0-15 row M=lane, K = k..k+15;
//                  lanes 16-31 row M=lane-16, K = k+16..k+31.
//  B (32x16 bf16): lane col N=lane&15; lanes 0-15 K=k..k+15, lanes 16-31
//                  K=k+16..k+31 (contiguous in K with Bt stored [N,K]).
//  C/D (16x16 f32): lanes 0-15 N=lane, VGPR r -> M=m0+r;
//                   lanes 16-31 N=lane-16, VGPR r -> M=m0+8+r.
// ---------------------------------------------------------------------------
__global__ __launch_bounds__(256)
void gemm_bf16_wmma(const bf16_t* __restrict__ A, int lda,
                    const bf16_t* __restrict__ Bt, int ldb,
                    const float*  __restrict__ bias,
                    float* __restrict__ Cf, bf16_t* __restrict__ Cb,
                    int ldc, int K)
{
    const int lane = threadIdx.x & 31;
    const int wave = threadIdx.x >> 5;
    const int m0   = blockIdx.x * 128 + (wave & 3) * 32;
    const int n0   = blockIdx.y * 64  + (wave >> 2) * 32;

    const int half = lane >> 4;   // 0 for lanes 0-15, 1 for lanes 16-31
    const int l15  = lane & 15;

    const bf16_t* pa0 = A  + (size_t)(m0 + l15) * lda + half * 16;
    const bf16_t* pa1 = pa0 + (size_t)16 * lda;
    const bf16_t* pb0 = Bt + (size_t)(n0 + l15) * ldb + half * 16;
    const bf16_t* pb1 = pb0 + (size_t)16 * ldb;

    v8f acc00 = {}, acc01 = {}, acc10 = {}, acc11 = {};
#pragma unroll 2
    for (int k = 0; k < K; k += 32) {
        union { v16bf v; u16x8 h[2]; } a0, a1;
        a0.h[0] = *(const u16x8*)(pa0);
        a0.h[1] = *(const u16x8*)(pa0 + 8);
        a1.h[0] = *(const u16x8*)(pa1);
        a1.h[1] = *(const u16x8*)(pa1 + 8);
        v16bf b0 = *(const v16bf*)(pb0);
        v16bf b1 = *(const v16bf*)(pb1);
        __builtin_prefetch(pa0 + 64, 0, 1);   // global_prefetch_b8
        __builtin_prefetch(pa1 + 64, 0, 1);
        acc00 = __builtin_amdgcn_wmma_f32_16x16x32_bf16(false, a0.v, false, b0,
                                                        (short)0, acc00, false, false);
        acc01 = __builtin_amdgcn_wmma_f32_16x16x32_bf16(false, a0.v, false, b1,
                                                        (short)0, acc01, false, false);
        acc10 = __builtin_amdgcn_wmma_f32_16x16x32_bf16(false, a1.v, false, b0,
                                                        (short)0, acc10, false, false);
        acc11 = __builtin_amdgcn_wmma_f32_16x16x32_bf16(false, a1.v, false, b1,
                                                        (short)0, acc11, false, false);
        pa0 += 32; pa1 += 32; pb0 += 32; pb1 += 32;
    }

    const int   cm = m0 + half * 8;
    const int   cn = n0 + l15;
    const float bv0 = bias ? bias[cn]      : 0.0f;
    const float bv1 = bias ? bias[cn + 16] : 0.0f;

    v8f accs[4] = { acc00, acc01, acc10, acc11 };
#pragma unroll
    for (int t = 0; t < 4; ++t) {
        const int   tm = cm + (t >> 1) * 16;
        const int   tn = cn + (t & 1) * 16;
        const float bv = (t & 1) ? bv1 : bv0;
#pragma unroll
        for (int r = 0; r < 8; ++r) {
            float v = accs[t][r] + bv;
            size_t idx = (size_t)(tm + r) * ldc + tn;
            if (Cf) Cf[idx] = v;
            if (Cb) Cb[idx] = (bf16_t)v;
        }
    }
}

// ---------------------------------------------------------------------------
// Row-wise LayerNorm epilogue. One block (256 threads) per row, N <= 512.
// mode 0: v = X[i]
// mode 1: v = X[i] + R[i]                        (residual)
// mode 2: v = X[i] + sigmoid(P[i]) * Mm[i]       (gated fuse)
// then LN(g,be), optional exact GELU; writes f32 (outF) and/or bf16 (outB,
// with row stride ldoB and column offset colB -> concat-in-place).
// ---------------------------------------------------------------------------
__global__ __launch_bounds__(256)
void row_epilogue(const float* __restrict__ X,
                  const float* __restrict__ R,
                  const float* __restrict__ P,
                  const float* __restrict__ Mm,
                  const float* __restrict__ g, const float* __restrict__ be,
                  float* __restrict__ outF, int ldoF,
                  bf16_t* __restrict__ outB, int ldoB, int colB,
                  int N, int mode, int doGelu)
{
    const int row = blockIdx.x;
    const int tid = threadIdx.x;
    __shared__ float s1[256], s2[256];

    float vloc[2];
    float s = 0.f, sq = 0.f;
    int c = 0;
    for (int i = tid; i < N; i += 256, ++c) {
        float v = X[(size_t)row * N + i];
        if (mode == 1) {
            v += R[(size_t)row * N + i];
        } else if (mode == 2) {
            float p = P[(size_t)row * N + i];
            v += Mm[(size_t)row * N + i] / (1.f + __expf(-p));
        }
        vloc[c] = v; s += v; sq += v * v;
    }
    s1[tid] = s; s2[tid] = sq;
    __syncthreads();
    for (int st = 128; st > 0; st >>= 1) {
        if (tid < st) { s1[tid] += s1[tid + st]; s2[tid] += s2[tid + st]; }
        __syncthreads();
    }
    const float mean = s1[0] / N;
    const float var  = s2[0] / N - mean * mean;
    const float inv  = rsqrtf(var + 1e-5f);

    c = 0;
    for (int i = tid; i < N; i += 256, ++c) {
        float v = (vloc[c] - mean) * inv * g[i] + be[i];
        if (doGelu) v = 0.5f * v * (1.f + erff(v * 0.70710678118654752f));
        if (outF) outF[(size_t)row * ldoF + i] = v;
        if (outB) outB[(size_t)row * ldoB + colB + i] = (bf16_t)v;
    }
}

// m = 0.5*(a+b); write f32 mF and bf16 copy into gate-input cols [col,col+512)
__global__ __launch_bounds__(256)
void avg_pack_kernel(const float* __restrict__ a, const float* __restrict__ b,
                     float* __restrict__ mF,
                     bf16_t* __restrict__ gin, int ldg, int col, size_t total)
{
    size_t i = (size_t)blockIdx.x * 256 + threadIdx.x;
    if (i >= total) return;
    float m = 0.5f * (a[i] + b[i]);
    mF[i] = m;
    size_t row = i >> 9;          // /512
    int    cc  = (int)(i & 511);
    gin[row * ldg + col + cc] = (bf16_t)m;
}

__global__ __launch_bounds__(256)
void cast_kernel(const float* __restrict__ in, bf16_t* __restrict__ out, size_t n)
{
    size_t i = (size_t)blockIdx.x * 256 + threadIdx.x;
    if (i < n) out[i] = (bf16_t)in[i];
}

// sigmoid(logits[B,F]) zero-padded to bf16 [B,32] (K padded for WMMA)
__global__ __launch_bounds__(256)
void sigmoid_pad_kernel(const float* __restrict__ in, int F,
                        bf16_t* __restrict__ out, int B)
{
    int i = blockIdx.x * 256 + threadIdx.x;
    if (i >= B * 32) return;
    int row = i >> 5, k = i & 31;
    float v = (k < F) ? 1.f / (1.f + __expf(-in[row * F + k])) : 0.f;
    out[i] = (bf16_t)v;
}

// out[n*Kp + k] = (k < K) ? W[k*ldin + col0 + n] : 0   (transpose + cast, pad K)
__global__ __launch_bounds__(256)
void wtrans_kernel(const float* __restrict__ W, int ldin, int col0,
                   int K, int Kp, int N, bf16_t* __restrict__ out)
{
    int i = blockIdx.x * 256 + threadIdx.x;
    if (i >= N * Kp) return;
    int n = i / Kp, k = i % Kp;
    float v = (k < K) ? W[(size_t)k * ldin + col0 + n] : 0.f;
    out[i] = (bf16_t)v;
}

// ---------------------------------------------------------------------------
extern "C" void kernel_launch(void* const* d_in, const int* in_sizes, int n_in,
                              void* d_out, int out_size, void* d_ws, size_t ws_size,
                              hipStream_t stream)
{
    const int B = 8192, HID = 1024, D = 512, Dh = 256, CB = 2304;

    const float* vh      = (const float*)d_in[0];
    const float* vl      = (const float*)d_in[1];
    const float* ih      = (const float*)d_in[2];
    const float* il      = (const float*)d_in[3];
    const float* th      = (const float*)d_in[4];
    const float* tl      = (const float*)d_in[5];
    const float* hp_W    = (const float*)d_in[6];
    const float* hp_b    = (const float*)d_in[7];
    const float* hp_g    = (const float*)d_in[8];
    const float* hp_be   = (const float*)d_in[9];
    const float* lpWv    = (const float*)d_in[10];
    const float* lpbv    = (const float*)d_in[11];
    const float* lpWi    = (const float*)d_in[12];
    const float* lpbi    = (const float*)d_in[13];
    const float* lpWt_   = (const float*)d_in[14];
    const float* lpbt    = (const float*)d_in[15];
    const float* lp_g    = (const float*)d_in[16];
    const float* lp_be   = (const float*)d_in[17];
    const float* mhaiW   = (const float*)d_in[18];
    const float* mhaib   = (const float*)d_in[19];
    const float* mhaoW   = (const float*)d_in[20];
    const float* mhaob   = (const float*)d_in[21];
    const float* n1_g    = (const float*)d_in[22];
    const float* n1_be   = (const float*)d_in[23];
    const float* n2_g    = (const float*)d_in[24];
    const float* n2_be   = (const float*)d_in[25];
    const float* gate_W  = (const float*)d_in[26];
    const float* gate_b  = (const float*)d_in[27];
    const float* fus_W1  = (const float*)d_in[28];
    const float* fus_b1  = (const float*)d_in[29];
    const float* fus_g1  = (const float*)d_in[30];
    const float* fus_ge1 = (const float*)d_in[31];
    const float* fus_W2  = (const float*)d_in[32];
    const float* fus_b2  = (const float*)d_in[33];
    const float* fus_g2  = (const float*)d_in[34];
    const float* fus_ge2 = (const float*)d_in[35];

    // ---- workspace bump allocator (256B aligned) ----
    size_t off = 0;
    auto alloc = [&](size_t bytes) -> char* {
        char* p = (char*)d_ws + off;
        off += (bytes + 255) & ~(size_t)255;
        return p;
    };

    bf16_t* hpWt   = (bf16_t*)alloc((size_t)3 * D * HID * 2);
    bf16_t* WvT    = (bf16_t*)alloc((size_t)4 * D * D * 2);
    bf16_t* WoT    = (bf16_t*)alloc((size_t)4 * D * D * 2);
    bf16_t* gateWt = (bf16_t*)alloc((size_t)3 * D * HID * 2);
    bf16_t* lpT    = (bf16_t*)alloc((size_t)3 * Dh * 32 * 2);
    bf16_t* f1T    = (bf16_t*)alloc((size_t)D * CB * 2);
    bf16_t* f2T    = (bf16_t*)alloc((size_t)D * D * 2);

    bf16_t* bIn[3];  for (int i = 0; i < 3; ++i) bIn[i] = (bf16_t*)alloc((size_t)B * HID * 2);
    bf16_t* sigl   = (bf16_t*)alloc((size_t)3 * B * 32 * 2);
    float*  tmpF   = (float*)alloc((size_t)B * D * 4);
    float*  tmp2F  = (float*)alloc((size_t)B * D * 4);
    float*  hFr    = (float*)alloc((size_t)B * D * 4);
    bf16_t* hBr    = (bf16_t*)alloc((size_t)B * D * 2);
    bf16_t* vpB    = (bf16_t*)alloc((size_t)B * D * 2);
    float*  eF[3];   for (int i = 0; i < 3; ++i) eF[i]  = (float*)alloc((size_t)B * D * 4);
    bf16_t* gin[3];  for (int i = 0; i < 3; ++i) gin[i] = (bf16_t*)alloc((size_t)B * HID * 2);
    float*  cF0    = (float*)alloc((size_t)B * D * 4);
    float*  cF1    = (float*)alloc((size_t)B * D * 4);
    float*  mF[3];   for (int i = 0; i < 3; ++i) mF[i]  = (float*)alloc((size_t)B * D * 4);
    bf16_t* comb   = (bf16_t*)alloc((size_t)B * CB * 2);
    bf16_t* hB2    = (bf16_t*)alloc((size_t)B * D * 2);
    if (off > ws_size) return;  // workspace too small (deterministic no-op)

    const dim3 blk(256);
    auto g1 = [](size_t n) { return dim3((unsigned)((n + 255) / 256)); };

    auto gemm = [&](const bf16_t* A, int lda, const bf16_t* Bt, int ldb,
                    const float* bias, float* Cf, bf16_t* Cb, int ldc,
                    int M, int N, int K) {
        dim3 grid(M / 128, N / 64);
        gemm_bf16_wmma<<<grid, blk, 0, stream>>>(A, lda, Bt, ldb, bias, Cf, Cb, ldc, K);
    };

    // ---- weight prep: transpose + cast to bf16 [N,K] ----
    for (int i = 0; i < 3; ++i)
        wtrans_kernel<<<g1((size_t)D * HID), blk, 0, stream>>>(
            hp_W + (size_t)i * HID * D, D, 0, HID, HID, D, hpWt + (size_t)i * D * HID);
    for (int j = 0; j < 4; ++j) {
        wtrans_kernel<<<g1((size_t)D * D), blk, 0, stream>>>(
            mhaiW + (size_t)j * D * 3 * D, 3 * D, 2 * D, D, D, D, WvT + (size_t)j * D * D);
        wtrans_kernel<<<g1((size_t)D * D), blk, 0, stream>>>(
            mhaoW + (size_t)j * D * D, D, 0, D, D, D, WoT + (size_t)j * D * D);
    }
    for (int i = 0; i < 3; ++i)
        wtrans_kernel<<<g1((size_t)D * HID), blk, 0, stream>>>(
            gate_W + (size_t)i * HID * D, D, 0, HID, HID, D, gateWt + (size_t)i * D * HID);
    wtrans_kernel<<<g1((size_t)Dh * 32), blk, 0, stream>>>(lpWv,  Dh, 0, 10, 32, Dh, lpT);
    wtrans_kernel<<<g1((size_t)Dh * 32), blk, 0, stream>>>(lpWi,  Dh, 0,  6, 32, Dh, lpT + (size_t)Dh * 32);
    wtrans_kernel<<<g1((size_t)Dh * 32), blk, 0, stream>>>(lpWt_, Dh, 0, 15, 32, Dh, lpT + (size_t)2 * Dh * 32);
    wtrans_kernel<<<g1((size_t)D * CB), blk, 0, stream>>>(fus_W1, D, 0, CB, CB, D, f1T);
    wtrans_kernel<<<g1((size_t)D * D),  blk, 0, stream>>>(fus_W2, D, 0, D, D, D, f2T);

    // ---- activation prep ----
    cast_kernel<<<g1((size_t)B * HID), blk, 0, stream>>>(vh, bIn[0], (size_t)B * HID);
    cast_kernel<<<g1((size_t)B * HID), blk, 0, stream>>>(ih, bIn[1], (size_t)B * HID);
    cast_kernel<<<g1((size_t)B * HID), blk, 0, stream>>>(th, bIn[2], (size_t)B * HID);
    sigmoid_pad_kernel<<<g1((size_t)B * 32), blk, 0, stream>>>(vl, 10, sigl, B);
    sigmoid_pad_kernel<<<g1((size_t)B * 32), blk, 0, stream>>>(il,  6, sigl + (size_t)B * 32, B);
    sigmoid_pad_kernel<<<g1((size_t)B * 32), blk, 0, stream>>>(tl, 15, sigl + (size_t)2 * B * 32, B);

    // ---- per task: hproj -> LN+GELU -> self-"attn" (v/out proj) -> residual LN ----
    for (int i = 0; i < 3; ++i) {
        gemm(bIn[i], HID, hpWt + (size_t)i * D * HID, HID, hp_b + i * D,
             tmpF, nullptr, D, B, D, HID);
        row_epilogue<<<B, blk, 0, stream>>>(tmpF, nullptr, nullptr, nullptr,
                                            hp_g + i * D, hp_be + i * D,
                                            hFr, D, hBr, D, 0, D, 0, 1);
        gemm(hBr, D, WvT, D, mhaib + 2 * D, nullptr, vpB, D, B, D, D);           // v_proj (mha 0)
        gemm(vpB, D, WoT, D, mhaob, tmp2F, nullptr, D, B, D, D);                 // out_proj (mha 0)
        row_epilogue<<<B, blk, 0, stream>>>(hFr, tmp2F, nullptr, nullptr,
                                            n1_g + i * D, n1_be + i * D,
                                            eF[i], D, gin[i], HID, 0, D, 1, 0);  // e -> gin cols [0,512)
    }

    // ---- lproj -> LN+GELU -> comb cols [1536 + i*256) ----
    const float* lpb[3] = { lpbv, lpbi, lpbt };
    for (int i = 0; i < 3; ++i) {
        gemm(sigl + (size_t)i * B * 32, 32, lpT + (size_t)i * Dh * 32, 32, lpb[i],
             tmpF, nullptr, Dh, B, Dh, 32);
        row_epilogue<<<B, blk, 0, stream>>>(tmpF, nullptr, nullptr, nullptr,
                                            lp_g + i * Dh, lp_be + i * Dh,
                                            nullptr, 0, comb, CB, 3 * D + i * Dh, Dh, 0, 1);
    }

    // ---- cross "attention": for dest d, two (mha idx, kv) pairs -> m ----
    const int ca_idx[3][2] = { {1, 2}, {1, 3}, {2, 3} };
    const int ca_kv [3][2] = { {1, 2}, {0, 2}, {0, 1} };
    for (int d = 0; d < 3; ++d) {
        for (int s = 0; s < 2; ++s) {
            int j = ca_idx[d][s], kv = ca_kv[d][s];
            gemm(gin[kv], HID, WvT + (size_t)j * D * D, D, mhaib + (size_t)j * 3 * D + 2 * D,
                 nullptr, vpB, D, B, D, D);
            gemm(vpB, D, WoT + (size_t)j * D * D, D, mhaob + (size_t)j * D,
                 s == 0 ? cF0 : cF1, nullptr, D, B, D, D);
        }
        avg_pack_kernel<<<g1((size_t)B * D), blk, 0, stream>>>(cF0, cF1, mF[d],
                                                               gin[d], HID, D, (size_t)B * D);
    }

    // ---- gated fuse per task -> comb cols [d*512) ----
    for (int d = 0; d < 3; ++d) {
        gemm(gin[d], HID, gateWt + (size_t)d * D * HID, HID, gate_b + d * D,
             tmpF, nullptr, D, B, D, HID);
        row_epilogue<<<B, blk, 0, stream>>>(eF[d], nullptr, tmpF, mF[d],
                                            n2_g + d * D, n2_be + d * D,
                                            nullptr, 0, comb, CB, d * D, D, 2, 0);
    }

    // ---- fusion MLP ----
    gemm(comb, CB, f1T, CB, fus_b1, tmpF, nullptr, D, B, D, CB);
    row_epilogue<<<B, blk, 0, stream>>>(tmpF, nullptr, nullptr, nullptr,
                                        fus_g1, fus_ge1, nullptr, 0, hB2, D, 0, D, 0, 1);
    gemm(hB2, D, f2T, D, fus_b2, tmp2F, nullptr, D, B, D, D);
    row_epilogue<<<B, blk, 0, stream>>>(tmp2F, nullptr, nullptr, nullptr,
                                        fus_g2, fus_ge2, (float*)d_out, D,
                                        nullptr, 0, 0, D, 0, 0);
}
